// NodeGATNet_9036611191478
// MI455X (gfx1250) — matrix operand order repaired and checked by reading, not compiled, not run on hardware
//
#include <hip/hip_runtime.h>
#include <hip/hip_bf16.h>

#define N_NODES 10000
#define NUM_E   160000
#define ETOT    (NUM_E + N_NODES)
#define F_IN    128
#define H1      10
#define C1      128
#define HC1     (H1 * C1)   /* 1280 */
#define C2      1024
#define OUTF    1024

typedef __attribute__((ext_vector_type(16))) __bf16 v16bf;
typedef __attribute__((ext_vector_type(8)))  float  v8f;

#if defined(__HIP_DEVICE_COMPILE__) && \
    __has_builtin(__builtin_amdgcn_global_load_async_to_lds_b128) && \
    __has_builtin(__builtin_amdgcn_s_wait_asynccnt)
#define USE_ASYNC_LDS 1
typedef int v4i __attribute__((ext_vector_type(4)));
typedef __attribute__((address_space(1))) v4i v4i_g;   // global-AS int4
typedef __attribute__((address_space(3))) v4i v4i_l;   // LDS-AS int4
#else
#define USE_ASYNC_LDS 0
#endif

// ---------------------------------------------------------------- helpers
__device__ __forceinline__ unsigned short f32_to_bf16_rne(float f) {
  unsigned u = __float_as_uint(f);
  unsigned r = u + 0x7fffu + ((u >> 16) & 1u);
  return (unsigned short)(r >> 16);
}
__device__ __forceinline__ unsigned enc_ord(float f) {
  unsigned b = __float_as_uint(f);
  return (b & 0x80000000u) ? ~b : (b | 0x80000000u);
}
__device__ __forceinline__ float dec_ord(unsigned u) {
  return (u & 0x80000000u) ? __uint_as_float(u & 0x7fffffffu) : __uint_as_float(~u);
}

// ---------------------------------------------------------------- conversions
__global__ void k_f32_to_bf16(const float* __restrict__ s, unsigned short* __restrict__ d, size_t n) {
  size_t i = (size_t)blockIdx.x * blockDim.x + threadIdx.x;
  size_t st = (size_t)gridDim.x * blockDim.x;
  for (; i < n; i += st) d[i] = f32_to_bf16_rne(s[i]);
}

// W is K x Nn row-major f32; Wt is Nn x K bf16 (column-major view of W)
__global__ void k_transpose_bf16(const float* __restrict__ w, unsigned short* __restrict__ wt,
                                 int K, int Nn) {
  size_t n = (size_t)K * Nn;
  size_t i = (size_t)blockIdx.x * blockDim.x + threadIdx.x;
  size_t st = (size_t)gridDim.x * blockDim.x;
  for (; i < n; i += st) {
    int r = (int)(i / Nn);
    int c = (int)(i % Nn);
    wt[(size_t)c * K + r] = f32_to_bf16_rne(w[i]);
  }
}

// ---------------------------------------------------------------- WMMA GEMM
// C[M,Nn] = A[M,K](bf16,row-major) * Bt[Nn,K](bf16 = col-major B) (+bias)(+relu)
// block = 256 threads = 8 waves; block tile 16 x 256; wave tile 16 x 32 (2 WMMAs/k-step).
// A tile (16x32 bf16 = 1KB) double-buffered in LDS via async global->LDS DMA.
__global__ __launch_bounds__(256)
void k_gemm_bf16(const unsigned short* __restrict__ A, const unsigned short* __restrict__ Bt,
                 const float* __restrict__ bias, float* __restrict__ C,
                 int M, int Nn, int K, int act) {
  __shared__ unsigned lds_a[2][256];       // two 16x32 bf16 A tiles
  const int lane = threadIdx.x & 31;
  const int wave = threadIdx.x >> 5;
  const int m0 = blockIdx.y << 4;
  const int n0 = (blockIdx.x << 8) + (wave << 5);
  const int mloc = lane & 15;
  const int hh = lane >> 4;
  const int nc = lane & 15;
  const int tid = threadIdx.x;

  union { v8f v; float f[8]; } acc0, acc1;
  acc0.v = (v8f){0.f, 0.f, 0.f, 0.f, 0.f, 0.f, 0.f, 0.f};
  acc1.v = acc0.v;

  const int T = K >> 5;                    // k-steps of 32

#if USE_ASYNC_LDS
  // staging lanes: threads 0..63, each moves one b128 (16x32 bf16 tile = 64 x 16B)
  const int srow = tid >> 2;               // 0..15
  const int schunk = tid & 3;              // 0..3  (4 x 16B per row)
  const char* gA = (const char*)A;
  char* lA = (char*)lds_a;

  auto stage = [&](int t, int buf) {
    if (tid < 64) {
      size_t goff = ((size_t)(m0 + srow) * K + ((size_t)t << 5)) * 2 + ((size_t)schunk << 4);
      __builtin_amdgcn_global_load_async_to_lds_b128(
          (v4i_g*)(gA + goff), (v4i_l*)(lA + (buf << 10) + (tid << 4)), 0, 0);
    }
  };

  stage(0, 0);
  __builtin_amdgcn_s_wait_asynccnt(0);
  __syncthreads();
#endif

  for (int t = 0; t < T; ++t) {
#if USE_ASYNC_LDS
    const int buf = t & 1;
    if (t + 1 < T) stage(t + 1, buf ^ 1);  // overlap next A tile DMA with compute
    const unsigned* ab = lds_a[buf];
#else
    // synchronous fallback staging: 256 threads, one dword (2 bf16) each
    lds_a[0][tid] =
        *(const unsigned*)(A + (size_t)(m0 + (tid >> 4)) * K + (t << 5) + ((tid & 15) << 1));
    __syncthreads();
    const unsigned* ab = lds_a[0];
#endif

    union { unsigned u[8]; v16bf v; } af, bf0, bf1;
#pragma unroll
    for (int v = 0; v < 8; ++v) {
      // A fragment (contiguous k, matches B): K = hh*16 + 2v -> dword hh*8 + v
      af.u[v] = ab[(mloc << 4) + (hh << 3) + v];
    }
    const unsigned* bp0 = (const unsigned*)(Bt + (size_t)(n0 + nc) * K + (t << 5));
    const unsigned* bp1 = (const unsigned*)(Bt + (size_t)(n0 + 16 + nc) * K + (t << 5));
#pragma unroll
    for (int v = 0; v < 8; ++v) {
      // B fragment (contiguous k, matches A): K = hh*16 + 2v -> dword hh*8 + v
      bf0.u[v] = bp0[(hh << 3) + v];
      bf1.u[v] = bp1[(hh << 3) + v];
    }
    acc0.v = __builtin_amdgcn_wmma_f32_16x16x32_bf16(
        false, af.v, false, bf0.v, (short)0, acc0.v, false, false);
    acc1.v = __builtin_amdgcn_wmma_f32_16x16x32_bf16(
        false, af.v, false, bf1.v, (short)0, acc1.v, false, false);

#if USE_ASYNC_LDS
    __builtin_amdgcn_s_wait_asynccnt(0);   // next tile landed in LDS
    __syncthreads();                       // publish; also fences reads of current buf
#else
    __syncthreads();
#endif
  }

  const float bv0 = bias ? bias[n0 + nc] : 0.f;
  const float bv1 = bias ? bias[n0 + 16 + nc] : 0.f;
#pragma unroll
  for (int r = 0; r < 8; ++r) {            // D: M = r + 8*hh, N = lane&15
    int row = m0 + r + (hh << 3);
    float v0 = acc0.f[r] + bv0;
    float v1 = acc1.f[r] + bv1;
    if (act == 1) { v0 = fmaxf(v0, 0.f); v1 = fmaxf(v1, 0.f); }
    C[(size_t)row * Nn + n0 + nc] = v0;
    C[(size_t)row * Nn + n0 + 16 + nc] = v1;
  }
}

// ---------------------------------------------------------------- attention scores
// one wave per (node, head): a_s = <h, att_src>, a_d = <h, att_dst>
__global__ void k_attn_score(const float* __restrict__ h, const float* __restrict__ atts,
                             const float* __restrict__ attd, float* __restrict__ as,
                             float* __restrict__ ad, int nn, int H, int Cc) {
  int w = (int)(((size_t)blockIdx.x * blockDim.x + threadIdx.x) >> 5);
  int lane = threadIdx.x & 31;
  if (w >= nn * H) return;
  int i = w / H, hd = w % H;
  const float* hp = h + ((size_t)i * H + hd) * Cc;
  const float* sp = atts + (size_t)hd * Cc;
  const float* dp = attd + (size_t)hd * Cc;
  float ss = 0.f, sd = 0.f;
  for (int c = lane; c < Cc; c += 32) {
    float v = hp[c];
    ss += v * sp[c];
    sd += v * dp[c];
  }
#pragma unroll
  for (int off = 16; off; off >>= 1) {
    ss += __shfl_down(ss, off, 32);
    sd += __shfl_down(sd, off, 32);
  }
  if (lane == 0) { as[w] = ss; ad[w] = sd; }
}

// ---------------------------------------------------------------- fills
__global__ void k_fill_u32(unsigned* __restrict__ p, unsigned v, size_t n) {
  size_t i = (size_t)blockIdx.x * blockDim.x + threadIdx.x;
  size_t st = (size_t)gridDim.x * blockDim.x;
  for (; i < n; i += st) p[i] = v;
}
__global__ void k_init_bias(float* __restrict__ o, const float* __restrict__ b, size_t n, int HC) {
  size_t i = (size_t)blockIdx.x * blockDim.x + threadIdx.x;
  size_t st = (size_t)gridDim.x * blockDim.x;
  for (; i < n; i += st) o[i] = b[i % HC];
}

// ---------------------------------------------------------------- segment softmax
__device__ __forceinline__ void edge_sd(const int* es, const int* ed, int e, int& s, int& d) {
  if (e < NUM_E) { s = es[e]; d = ed[e]; }
  else           { s = e - NUM_E; d = s; }        // self-loops
}
__device__ __forceinline__ float edge_logit(const float* as, const float* ad,
                                            int s, int d, int hd, int H) {
  float x = as[s * H + hd] + ad[d * H + hd];
  return x > 0.f ? x : 0.2f * x;                   // leaky_relu(0.2)
}

__global__ void k_edge_max(const int* __restrict__ es, const int* __restrict__ ed,
                           const float* __restrict__ as, const float* __restrict__ ad,
                           unsigned* __restrict__ menc, int H) {
  size_t n = (size_t)ETOT * H;
  size_t i = (size_t)blockIdx.x * blockDim.x + threadIdx.x;
  size_t st = (size_t)gridDim.x * blockDim.x;
  for (; i < n; i += st) {
    int e = (int)(i / H), hd = (int)(i % H), s, d;
    edge_sd(es, ed, e, s, d);
    atomicMax(&menc[d * H + hd], enc_ord(edge_logit(as, ad, s, d, hd, H)));
  }
}
__global__ void k_edge_sum(const int* __restrict__ es, const int* __restrict__ ed,
                           const float* __restrict__ as, const float* __restrict__ ad,
                           const unsigned* __restrict__ menc, float* __restrict__ ssum, int H) {
  size_t n = (size_t)ETOT * H;
  size_t i = (size_t)blockIdx.x * blockDim.x + threadIdx.x;
  size_t st = (size_t)gridDim.x * blockDim.x;
  for (; i < n; i += st) {
    int e = (int)(i / H), hd = (int)(i % H), s, d;
    edge_sd(es, ed, e, s, d);
    float x = edge_logit(as, ad, s, d, hd, H);
    atomicAdd(&ssum[d * H + hd], __expf(x - dec_ord(menc[d * H + hd])));
  }
}
__global__ void k_edge_alpha(const int* __restrict__ es, const int* __restrict__ ed,
                             const float* __restrict__ as, const float* __restrict__ ad,
                             const unsigned* __restrict__ menc, const float* __restrict__ ssum,
                             float* __restrict__ alpha, int H) {
  size_t n = (size_t)ETOT * H;
  size_t i = (size_t)blockIdx.x * blockDim.x + threadIdx.x;
  size_t st = (size_t)gridDim.x * blockDim.x;
  for (; i < n; i += st) {
    int e = (int)(i / H), hd = (int)(i % H), s, d;
    edge_sd(es, ed, e, s, d);
    float x = edge_logit(as, ad, s, d, hd, H);
    float ex = __expf(x - dec_ord(menc[d * H + hd]));
    alpha[i] = ex / (ssum[d * H + hd] + 1e-16f);
  }
}

// ---------------------------------------------------------------- scatter-add aggregation
__global__ void k_aggregate(const int* __restrict__ es, const int* __restrict__ ed,
                            const float* __restrict__ h, const float* __restrict__ alpha,
                            float* __restrict__ out, int H, int Cc) {
  size_t n = (size_t)ETOT * H * Cc;
  size_t i = (size_t)blockIdx.x * blockDim.x + threadIdx.x;
  size_t st = (size_t)gridDim.x * blockDim.x;
  for (; i < n; i += st) {
    int c = (int)(i % Cc);
    size_t t = i / Cc;
    int hd = (int)(t % H);
    int e = (int)(t / H), s, d;
    edge_sd(es, ed, e, s, d);
    float v = h[((size_t)s * H + hd) * Cc + c] * alpha[t];
    atomicAdd(&out[((size_t)d * H + hd) * Cc + c], v);
  }
}

// ---------------------------------------------------------------- activation -> bf16
__global__ void k_act_bf16(const float* __restrict__ s, unsigned short* __restrict__ d,
                           size_t n, int mode) {
  size_t i = (size_t)blockIdx.x * blockDim.x + threadIdx.x;
  size_t st = (size_t)gridDim.x * blockDim.x;
  for (; i < n; i += st) {
    float v = s[i];
    if (mode == 0) v = (v > 0.f) ? v : (__expf(v) - 1.f);  // ELU
    else           v = fmaxf(v, 0.f);                      // ReLU
    d[i] = f32_to_bf16_rne(v);
  }
}

// ---------------------------------------------------------------- launch
extern "C" void kernel_launch(void* const* d_in, const int* in_sizes, int n_in,
                              void* d_out, int out_size, void* d_ws, size_t ws_size,
                              hipStream_t stream) {
  const float* node = (const float*)d_in[0];
  const int*   ei   = (const int*)d_in[1];
  const float* W1   = (const float*)d_in[2];
  const float* atts1 = (const float*)d_in[3];
  const float* attd1 = (const float*)d_in[4];
  const float* b1   = (const float*)d_in[5];
  const float* W2   = (const float*)d_in[6];
  const float* atts2 = (const float*)d_in[7];
  const float* attd2 = (const float*)d_in[8];
  const float* b2   = (const float*)d_in[9];
  const float* fcw  = (const float*)d_in[10];
  const float* fcb  = (const float*)d_in[11];
  float* out = (float*)d_out;
  char* ws = (char*)d_ws;

  size_t off = 0;
  auto take = [&](size_t bytes) -> char* {
    char* p = ws + off;
    off += (bytes + 255) & ~(size_t)255;
    return p;
  };
  unsigned short* node_bf = (unsigned short*)take((size_t)N_NODES * F_IN * 2);
  unsigned short* W1t  = (unsigned short*)take((size_t)HC1 * F_IN * 2);
  unsigned short* W2t  = (unsigned short*)take((size_t)C2 * HC1 * 2);
  unsigned short* fcwt = (unsigned short*)take((size_t)OUTF * OUTF * 2);
  float* hbuf   = (float*)take((size_t)N_NODES * HC1 * 4);   // h1, later h2
  float* aggbuf = (float*)take((size_t)N_NODES * HC1 * 4);   // out1, later out2
  float* as     = (float*)take((size_t)N_NODES * H1 * 4);
  float* ad     = (float*)take((size_t)N_NODES * H1 * 4);
  unsigned* menc = (unsigned*)take((size_t)N_NODES * H1 * 4);
  float* ssum   = (float*)take((size_t)N_NODES * H1 * 4);
  float* alpha  = (float*)take((size_t)ETOT * H1 * 4);
  unsigned short* x1bf = (unsigned short*)take((size_t)N_NODES * HC1 * 2);
  unsigned short* x2bf = (unsigned short*)take((size_t)N_NODES * C2 * 2);
  if (off > ws_size) return;

  const int* esrc = ei;
  const int* edst = ei + NUM_E;
  auto nb = [](size_t n) -> unsigned {
    size_t b = (n + 255) / 256;
    return (unsigned)(b > (1u << 20) ? (1u << 20) : b);
  };

  // ---- input conversions
  k_f32_to_bf16<<<nb((size_t)N_NODES * F_IN), 256, 0, stream>>>(node, node_bf, (size_t)N_NODES * F_IN);
  k_transpose_bf16<<<nb((size_t)F_IN * HC1), 256, 0, stream>>>(W1, W1t, F_IN, HC1);
  k_transpose_bf16<<<nb((size_t)HC1 * C2), 256, 0, stream>>>(W2, W2t, HC1, C2);
  k_transpose_bf16<<<nb((size_t)OUTF * OUTF), 256, 0, stream>>>(fcw, fcwt, OUTF, OUTF);

  // ---- layer 1: h1 = node @ W1
  dim3 g1(HC1 / 256, N_NODES / 16);
  k_gemm_bf16<<<g1, 256, 0, stream>>>(node_bf, W1t, nullptr, hbuf, N_NODES, HC1, F_IN, 0);
  k_attn_score<<<nb((size_t)N_NODES * H1 * 32), 256, 0, stream>>>(hbuf, atts1, attd1, as, ad, N_NODES, H1, C1);
  k_fill_u32<<<nb((size_t)N_NODES * H1), 256, 0, stream>>>(menc, 0u, (size_t)N_NODES * H1);
  k_fill_u32<<<nb((size_t)N_NODES * H1), 256, 0, stream>>>((unsigned*)ssum, 0u, (size_t)N_NODES * H1);
  k_edge_max<<<nb((size_t)ETOT * H1), 256, 0, stream>>>(esrc, edst, as, ad, menc, H1);
  k_edge_sum<<<nb((size_t)ETOT * H1), 256, 0, stream>>>(esrc, edst, as, ad, menc, ssum, H1);
  k_edge_alpha<<<nb((size_t)ETOT * H1), 256, 0, stream>>>(esrc, edst, as, ad, menc, ssum, alpha, H1);
  k_init_bias<<<nb((size_t)N_NODES * HC1), 256, 0, stream>>>(aggbuf, b1, (size_t)N_NODES * HC1, HC1);
  k_aggregate<<<nb((size_t)ETOT * H1 * C1), 256, 0, stream>>>(esrc, edst, hbuf, alpha, aggbuf, H1, C1);
  k_act_bf16<<<nb((size_t)N_NODES * HC1), 256, 0, stream>>>(aggbuf, x1bf, (size_t)N_NODES * HC1, 0);

  // ---- layer 2: h2 = x1 @ W2 (reuse hbuf/aggbuf)
  dim3 g2(C2 / 256, N_NODES / 16);
  k_gemm_bf16<<<g2, 256, 0, stream>>>(x1bf, W2t, nullptr, hbuf, N_NODES, C2, HC1, 0);
  k_attn_score<<<nb((size_t)N_NODES * 32), 256, 0, stream>>>(hbuf, atts2, attd2, as, ad, N_NODES, 1, C2);
  k_fill_u32<<<nb((size_t)N_NODES), 256, 0, stream>>>(menc, 0u, (size_t)N_NODES);
  k_fill_u32<<<nb((size_t)N_NODES), 256, 0, stream>>>((unsigned*)ssum, 0u, (size_t)N_NODES);
  k_edge_max<<<nb((size_t)ETOT), 256, 0, stream>>>(esrc, edst, as, ad, menc, 1);
  k_edge_sum<<<nb((size_t)ETOT), 256, 0, stream>>>(esrc, edst, as, ad, menc, ssum, 1);
  k_edge_alpha<<<nb((size_t)ETOT), 256, 0, stream>>>(esrc, edst, as, ad, menc, ssum, alpha, 1);
  k_init_bias<<<nb((size_t)N_NODES * C2), 256, 0, stream>>>(aggbuf, b2, (size_t)N_NODES * C2, C2);
  k_aggregate<<<nb((size_t)ETOT * C2), 256, 0, stream>>>(esrc, edst, hbuf, alpha, aggbuf, 1, C2);
  k_act_bf16<<<nb((size_t)N_NODES * C2), 256, 0, stream>>>(aggbuf, x2bf, (size_t)N_NODES * C2, 1);

  // ---- fc: out = relu(x2 @ fc_w + fc_b)
  dim3 g3(OUTF / 256, N_NODES / 16);
  k_gemm_bf16<<<g3, 256, 0, stream>>>(x2bf, fcwt, fcb, out, N_NODES, OUTF, OUTF, 1);
}
